// WindowAttention_68238440399251
// MI455X (gfx1250) — matrix operand order repaired and checked by reading, not compiled, not run on hardware
//
#include <hip/hip_runtime.h>

// ---------------------------------------------------------------------------
// Fused Swin window attention for MI455X (gfx1250, wave32, WMMA).
//   B=4096 windows, N=49 tokens (padded to 64), C=384, H=12 heads, d=32.
// One workgroup (256 threads = 8 wave32) processes one window end-to-end:
//   x(f32)->LDS bf16 -> QKV wmma -> QK^T wmma -> softmax(+rel bias) ->
//   PV wmma -> projection wmma -> f32 out.
// All matrix math: v_wmma_f32_16x16x32_bf16 (f32 accumulation).
// Weights pre-converted to bf16 in d_ws, served from the 192 MB L2.
// D-fragments are column-contiguous per lane (8 consecutive M at fixed N),
// so transposed outputs (V^T, S^T) are written with packed b128 stores.
// ---------------------------------------------------------------------------

#define DIMC   384
#define NTOK   49
#define HEADS  12
#define HD     32
#define NPAD   64
#define KSTEPS 12            // 384 / 32

typedef __bf16 bf16;
typedef __attribute__((ext_vector_type(16))) __bf16 v16bf;
typedef __attribute__((ext_vector_type(8)))  __bf16 bf16x8;
typedef __attribute__((ext_vector_type(4)))  __bf16 bf16x4;
typedef __attribute__((ext_vector_type(8)))  float  v8f;

// LDS layout (bytes), all 16B aligned; total 141,824 B (< 320 KB WGP LDS)
#define OFF_XBF   0        // [64][392] bf16  x tile (row-major, padded)
#define OFF_OBF   50176    // [64][392] bf16  attention output (proj input)
#define OFF_QS    100352   // [64][40]  bf16  Q (scaled), row-major
#define OFF_KS    105472   // [64][40]  bf16  K, row-major
#define OFF_VT    110592   // [32][72]  bf16  V transposed (d-major)
#define OFF_S     115200   // [64][68]  f32   scores, TRANSPOSED: St[j][i]
#define OFF_P     132608   // [64][72]  bf16  softmax probs, row-major
#define SMEM_BYTES 141824
#define XS 392
#define QKS 40
#define VS 72
#define SS 68
#define PS 72

__device__ __forceinline__ v8f wmma_bf16(v16bf a, v16bf b, v8f c) {
  return __builtin_amdgcn_wmma_f32_16x16x32_bf16(
      /*neg_a=*/false, a, /*neg_b=*/false, b,
      /*c_mod=*/(short)0, c, /*reuse_a=*/false, /*reuse_b=*/false);
}

// A fragment (16x32 bf16, row-major source). Per lane: m = lane&15,
// half = lane>>4; elements k = kbase + half*8 + {0..7} and +16 + {0..7}.
template <typename PT>
__device__ __forceinline__ v16bf load_frag_a(const PT* base, int stride,
                                             int mbase, int kbase, int lane) {
  int ml = lane & 15, half = lane >> 4;
  const PT* row = base + (size_t)(mbase + ml) * stride + kbase + half * 8;
  bf16x8 lo = *(const bf16x8*)(row);
  bf16x8 hi = *(const bf16x8*)(row + 16);
  v16bf a;
#pragma unroll
  for (int i = 0; i < 8; ++i) { a[i] = lo[i]; a[i + 8] = hi[i]; }
  return a;
}

// B fragment (32x16 bf16). Source stored "output-channel-major" (B^T row-major):
// per lane: n = lane&15, half = lane>>4; elements k = kbase + half*8 + {0..7} and +16 + {0..7}.
template <typename PT>
__device__ __forceinline__ v16bf load_frag_b(const PT* base, int stride,
                                             int nbase, int kbase, int lane) {
  int nl = lane & 15, half = lane >> 4;
  const PT* row = base + (size_t)(nbase + nl) * stride + kbase + half * 8;
  bf16x8 lo = *(const bf16x8*)(row);
  bf16x8 hi = *(const bf16x8*)(row + 16);
  v16bf b;
#pragma unroll
  for (int i = 0; i < 8; ++i) { b[i] = lo[i]; b[i + 8] = hi[i]; }
  return b;
}

__global__ void convert_weights_kernel(const float* __restrict__ qkv_w,
                                       const float* __restrict__ proj_w,
                                       bf16* __restrict__ qkv_wb,
                                       bf16* __restrict__ proj_wb) {
  const int n1 = 3 * DIMC * DIMC;
  const int n2 = DIMC * DIMC;
  for (int i = blockIdx.x * blockDim.x + threadIdx.x; i < n1 + n2;
       i += gridDim.x * blockDim.x) {
    if (i < n1) qkv_wb[i] = (bf16)qkv_w[i];
    else        proj_wb[i - n1] = (bf16)proj_w[i - n1];
  }
}

__global__ void __launch_bounds__(256)
win_attn_kernel(const float* __restrict__ x,
                const bf16* __restrict__ qkv_wb,
                const float* __restrict__ qkv_b,
                const bf16* __restrict__ proj_wb,
                const float* __restrict__ proj_b,
                const float* __restrict__ bias_table,
                const int* __restrict__ rel_idx,
                float* __restrict__ out) {
  extern __shared__ char smem[];
  bf16*  xbf = (bf16*)(smem + OFF_XBF);
  bf16*  obf = (bf16*)(smem + OFF_OBF);
  bf16*  Qs  = (bf16*)(smem + OFF_QS);
  bf16*  Ks  = (bf16*)(smem + OFF_KS);
  bf16*  Vt  = (bf16*)(smem + OFF_VT);
  float* St  = (float*)(smem + OFF_S);    // transposed scores: St[j*SS + i]
  bf16*  P   = (bf16*)(smem + OFF_P);

  const int b    = blockIdx.x;
  const int tid  = threadIdx.x;
  const int lane = tid & 31;
  const int nl   = lane & 15;
  const int half = lane >> 4;
  const int wave = tid >> 5;
  const float scale = 0.17677669529663687f;  // 32^-0.5

  __builtin_prefetch(qkv_wb, 0, 1);    // global_prefetch_b8 into cache
  __builtin_prefetch(proj_wb, 0, 1);

  // ---- stage x tile: f32 global -> bf16 LDS (vectorized) ----
  const float4* xg4 = (const float4*)(x + (size_t)b * NTOK * DIMC);
  for (int i = tid; i < NTOK * (DIMC / 4); i += 256) {
    int r = i / (DIMC / 4), c4 = i - r * (DIMC / 4);
    float4 v = xg4[i];
    bf16x4 pk = {(bf16)v.x, (bf16)v.y, (bf16)v.z, (bf16)v.w};
    *(bf16x4*)(&xbf[r * XS + c4 * 4]) = pk;
  }
  // zero-pad rows 49..63 (includes pad columns; harmless)
  {
    bf16x4 z = {(bf16)0.f, (bf16)0.f, (bf16)0.f, (bf16)0.f};
    bf16* padp = xbf + NTOK * XS;
    for (int i = tid; i < (NPAD - NTOK) * XS / 4; i += 256)
      *(bf16x4*)(padp + i * 4) = z;
  }
  __syncthreads();

  for (int h = 0; h < HEADS; ++h) {
    // ---- QKV GEMM: wave -> fixed m-tile, 3 n-tiles; A reused per k-step ----
    // n-tiles 0,1 -> Q ; 2,3 -> K ; 4,5 -> V
    {
      int mt = wave & 3;
      int nt0 = (wave >> 2) * 3;                    // 0 or 3
      v8f acc0 = {0.f,0.f,0.f,0.f,0.f,0.f,0.f,0.f};
      v8f acc1 = acc0, acc2 = acc0;
#pragma unroll 4
      for (int kk = 0; kk < KSTEPS; ++kk) {
        v16bf a = load_frag_a(xbf, XS, mt * 16, kk * 32, lane);
#pragma unroll
        for (int s = 0; s < 3; ++s) {
          int nt = nt0 + s;
          int ch_base = (nt >> 1) * DIMC + h * HD + (nt & 1) * 16;
          v16bf w = load_frag_b(qkv_wb, DIMC, ch_base, kk * 32, lane);
          if (s == 0)      acc0 = wmma_bf16(a, w, acc0);
          else if (s == 1) acc1 = wmma_bf16(a, w, acc1);
          else             acc2 = wmma_bf16(a, w, acc2);
        }
      }
#pragma unroll
      for (int s = 0; s < 3; ++s) {
        int nt = nt0 + s;
        int which = nt >> 1;
        int ch = which * DIMC + h * HD + (nt & 1) * 16 + nl;
        int dd = (nt & 1) * 16 + nl;
        float bias = qkv_b[ch];
        v8f acc = (s == 0) ? acc0 : (s == 1) ? acc1 : acc2;
        if (which == 2) {
          // V: transposed store, column-contiguous -> one b128
          bf16x8 pk;
#pragma unroll
          for (int r = 0; r < 8; ++r) pk[r] = (bf16)(acc[r] + bias);
          *(bf16x8*)(&Vt[dd * VS + mt * 16 + half * 8]) = pk;
        } else {
#pragma unroll
          for (int r = 0; r < 8; ++r) {
            int m = mt * 16 + r + half * 8;
            float v = acc[r] + bias;
            if (which == 0) Qs[m * QKS + dd] = (bf16)(v * scale);
            else            Ks[m * QKS + dd] = (bf16)v;
          }
        }
      }
    }
    __syncthreads();

    // ---- S = Q @ K^T : wave -> fixed m-tile, 2 n-tiles, K=32 (1 step) ----
    // D stored TRANSPOSED (St[j][i]) -> column-contiguous packed f32 stores.
    {
      int mt = wave & 3;
      int nt0 = (wave >> 2) * 2;
      v16bf a = load_frag_a(Qs, QKS, mt * 16, 0, lane);
#pragma unroll
      for (int s = 0; s < 2; ++s) {
        int nt = nt0 + s;
        v16bf kb = load_frag_b(Ks, QKS, nt * 16, 0, lane);
        v8f acc = {0.f,0.f,0.f,0.f,0.f,0.f,0.f,0.f};
        acc = wmma_bf16(a, kb, acc);
        float* dst = &St[(nt * 16 + nl) * SS + mt * 16 + half * 8];
        float4 lo = {acc[0], acc[1], acc[2], acc[3]};
        float4 hi = {acc[4], acc[5], acc[6], acc[7]};
        *(float4*)(dst)     = lo;
        *(float4*)(dst + 4) = hi;
      }
    }
    __syncthreads();

    // ---- softmax rows (+relative position bias); cols>=49 -> 0 ----
    // St is transposed: thread i reads St[j*SS + i] (conflict-free banks).
    if (tid < NPAD) {
      int i  = tid;
      int ii = (i < NTOK) ? i : (NTOK - 1);           // guard pad rows
      const int* ri = rel_idx + ii * NTOK;
      float mx = -3.0e38f;
      for (int j = 0; j < NTOK; ++j) {
        float sv = St[j * SS + i] + bias_table[ri[j] * HEADS + h];
        St[j * SS + i] = sv;
        mx = fmaxf(mx, sv);
      }
      float sum = 0.f;
      for (int j = 0; j < NTOK; ++j) {
        float e = __expf(St[j * SS + i] - mx);
        St[j * SS + i] = e;
        sum += e;
      }
      float rs = 1.f / sum;
      for (int j = 0; j < NTOK; ++j) P[i * PS + j] = (bf16)(St[j * SS + i] * rs);
      for (int j = NTOK; j < NPAD; ++j) P[i * PS + j] = (bf16)0.f;
    }
    __syncthreads();

    // ---- O_h = P @ V : 8 tiles (4 m x 2 n), K=64 (2 steps) ----
    {
      int mt = wave >> 1, nt = wave & 1;
      v8f acc = {0.f,0.f,0.f,0.f,0.f,0.f,0.f,0.f};
#pragma unroll
      for (int kk = 0; kk < 2; ++kk) {
        v16bf a  = load_frag_a(P, PS, mt * 16, kk * 32, lane);
        v16bf vb = load_frag_b(Vt, VS, nt * 16, kk * 32, lane);
        acc = wmma_bf16(a, vb, acc);
      }
      int col = h * HD + nt * 16 + nl;
#pragma unroll
      for (int r = 0; r < 8; ++r)
        obf[(mt * 16 + r + half * 8) * XS + col] = (bf16)acc[r];
    }
    __syncthreads();
  }

  // ---- projection: Y = O @ projW^T + proj_b ----
  // wave -> fixed m-tile, 12 n-tiles in 3 groups of 4 (A reused per k-step)
  float* og = out + (size_t)b * NTOK * DIMC;
  {
    int mt = wave & 3;
    int nb = (wave >> 2) * 12;
    for (int g = 0; g < 3; ++g) {
      v8f acc0 = {0.f,0.f,0.f,0.f,0.f,0.f,0.f,0.f};
      v8f acc1 = acc0, acc2 = acc0, acc3 = acc0;
#pragma unroll 4
      for (int kk = 0; kk < KSTEPS; ++kk) {
        v16bf a = load_frag_a(obf, XS, mt * 16, kk * 32, lane);
#pragma unroll
        for (int t = 0; t < 4; ++t) {
          int nt = nb + g * 4 + t;
          v16bf w = load_frag_b(proj_wb, DIMC, nt * 16, kk * 32, lane);
          if (t == 0)      acc0 = wmma_bf16(a, w, acc0);
          else if (t == 1) acc1 = wmma_bf16(a, w, acc1);
          else if (t == 2) acc2 = wmma_bf16(a, w, acc2);
          else             acc3 = wmma_bf16(a, w, acc3);
        }
      }
#pragma unroll
      for (int t = 0; t < 4; ++t) {
        int nt = nb + g * 4 + t;
        int col = nt * 16 + nl;
        float pb = proj_b[col];
        v8f acc = (t == 0) ? acc0 : (t == 1) ? acc1 : (t == 2) ? acc2 : acc3;
#pragma unroll
        for (int r = 0; r < 8; ++r) {
          int m = mt * 16 + r + half * 8;
          if (m < NTOK) og[(size_t)m * DIMC + col] = acc[r] + pb;
        }
      }
    }
  }
}

extern "C" void kernel_launch(void* const* d_in, const int* in_sizes, int n_in,
                              void* d_out, int out_size, void* d_ws, size_t ws_size,
                              hipStream_t stream) {
  const float* x          = (const float*)d_in[0];
  const float* qkv_w      = (const float*)d_in[1];
  const float* qkv_b      = (const float*)d_in[2];
  const float* proj_w     = (const float*)d_in[3];
  const float* proj_b     = (const float*)d_in[4];
  const float* bias_table = (const float*)d_in[5];
  const int*   rel_idx    = (const int*)d_in[6];

  const int B = in_sizes[0] / (NTOK * DIMC);   // 4096

  bf16* qkv_wb  = (bf16*)d_ws;
  bf16* proj_wb = (bf16*)((char*)d_ws + (size_t)3 * DIMC * DIMC * sizeof(bf16));

  convert_weights_kernel<<<512, 256, 0, stream>>>(qkv_w, proj_w, qkv_wb, proj_wb);

  win_attn_kernel<<<B, 256, SMEM_BYTES, stream>>>(
      x, qkv_wb, qkv_b, proj_wb, proj_b, bias_table, rel_idx, (float*)d_out);
}
